// RNN_53137335386692
// MI455X (gfx1250) — matrix operand order repaired and checked by reading, not compiled, not run on hardware
//
#include <hip/hip_runtime.h>

typedef __attribute__((ext_vector_type(16))) _Float16 v16h;
typedef __attribute__((ext_vector_type(8)))  _Float16 v8h;
typedef __attribute__((ext_vector_type(8)))  float    v8f;

#define KWIN   16
#define BB     8
#define LL     2048
#define DD     256
#define G3     768            // 3*h
#define LPAD   2064           // (l + K - 1) rounded to 16
#define LNEPS  1e-5f

union AFrag { v16h v; v8h h[2]; };

// ---------------------------------------------------------------------------
// f32 -> f16 weight conversion (keeps natural [n][k] row-major layout, which
// makes every WMMA B-fragment a contiguous 32-byte per-lane load).
__global__ void __launch_bounds__(256) cvt_w_kernel(const float* __restrict__ w,
                                                    _Float16* __restrict__ o, int n) {
    int i = blockIdx.x * blockDim.x + threadIdx.x;
    if (i < n) o[i] = (_Float16)w[i];
}

// Build zero-left-padded f16 copy of x: xpad[b][p][c], p in [0,2064),
// rows p<15 and p==2063 are zero; p in [15,2063) maps to x[b][p-15][c].
__global__ void __launch_bounds__(256) build_xpad_kernel(const float* __restrict__ x,
                                                         _Float16* __restrict__ xp) {
    int i = blockIdx.x * blockDim.x + threadIdx.x;      // over 8*2064*256
    int c = i & (DD - 1);
    int p = (i >> 8) % LPAD;
    int b = i / (LPAD * DD);
    float v = 0.0f;
    if (p >= (KWIN - 1) && p < (LL + KWIN - 1))
        v = x[((size_t)b * LL + (p - (KWIN - 1))) * DD + c];
    xp[i] = (_Float16)v;
}

// ---------------------------------------------------------------------------
// gi = xpad @ W_ih^T + b_ih   (M = 8*2064, N = 768, K = 256), f32 output.
// Output layout is channel-major interleaved: gi[row][chan*3 + gate], so the
// recurrence kernel reads r/z/n for one element as 12 contiguous bytes.
// Block: 256 threads = 8 waves; block M-tile = 64; each wave owns 6 N-tiles.
__global__ void __launch_bounds__(256) gi_gemm_kernel(const _Float16* __restrict__ xp,
                                                      const _Float16* __restrict__ wih,
                                                      const float* __restrict__ b_ih,
                                                      float* __restrict__ gi) {
    const int lane  = threadIdx.x & 31;
    const int w     = threadIdx.x >> 5;
    const int lh    = lane & 15;          // A: row-in-tile  B/C: col-in-tile
    const int hi    = (lane >> 4) & 1;    // which 16-lane half
    const int mbase = blockIdx.x * 64;

    for (int nt = 0; nt < 6; ++nt) {
        const int col  = (w * 6 + nt) * 16 + lh;        // gate-major column
        const int gate = col >> 8;
        const int chan = col & (DD - 1);
        const size_t csto = (size_t)chan * 3 + gate;    // interleaved offset
        const _Float16* bp = wih + (size_t)col * DD + hi * 16;
        v16h bf[8];
#pragma unroll
        for (int kc = 0; kc < 8; ++kc)
            bf[kc] = *(const v16h*)(bp + kc * 32);
        const float bias = b_ih[col];

#pragma unroll
        for (int mt = 0; mt < 4; ++mt) {
            const int m = mbase + mt * 16 + lh;
            const _Float16* ap = xp + (size_t)m * DD + hi * 16;
            v8f acc = {};
#pragma unroll
            for (int kc = 0; kc < 8; ++kc) {
                AFrag a;
                a.h[0] = *(const v8h*)(ap + kc * 32);
                a.h[1] = *(const v8h*)(ap + kc * 32 + 8);
                acc = __builtin_amdgcn_wmma_f32_16x16x32_f16(
                        false, a.v, false, bf[kc], (short)0, acc, false, false);
            }
            const int rb = mbase + mt * 16 + hi * 8;
#pragma unroll
            for (int i = 0; i < 8; ++i)
                gi[(size_t)(rb + i) * G3 + csto] = acc[i] + bias;
        }
    }
}

// ---------------------------------------------------------------------------
// 16-step GRU recurrence + residual LayerNorm.
// One block = 64 consecutive positions of one batch row. 8 waves.
// Wave w owns channels [32w, 32w+32): it computes the r/z/n gate tiles for
// those channels (N-tiles at c, 256+c, 512+c) for all 64 rows, so all three
// gate accumulators of an element live in the same wave's VGPRs.
// ALL W_hh B-fragments (384 VGPRs) are hoisted into registers before the
// recurrence: the 16-step inner loop is pure LDS A-loads + WMMA.
__global__ void __launch_bounds__(256) gru_ln_kernel(const _Float16* __restrict__ whh,
                                                     const float* __restrict__ gi,
                                                     const float* __restrict__ x,
                                                     const float* __restrict__ b_hh,
                                                     const float* __restrict__ gamma,
                                                     const float* __restrict__ beta,
                                                     float* __restrict__ out) {
    __shared__ _Float16 Hh[64][DD + 8];     // f16 state tile feeding WMMA A
    __shared__ float    rsum[64], rsumsq[64];

    const int lane = threadIdx.x & 31;
    const int w    = threadIdx.x >> 5;
    const int lh   = lane & 15;
    const int hi   = (lane >> 4) & 1;
    const int bi   = blockIdx.x >> 5;               // 32 tiles per batch
    const int t0   = (blockIdx.x & 31) * 64;
    const int cb   = w * 32;                        // wave channel base
    const float* gib = gi + (size_t)bi * LPAD * G3;

    for (int idx = threadIdx.x; idx < 64 * (DD + 8); idx += 256)
        (&Hh[0][0])[idx] = (_Float16)0.0f;
    if (threadIdx.x < 64) { rsum[threadIdx.x] = 0.0f; rsumsq[threadIdx.x] = 0.0f; }

    float bh[2][3];
#pragma unroll
    for (int ct = 0; ct < 2; ++ct) {
        const int ch = cb + ct * 16 + lh;
        bh[ct][0] = b_hh[ch];
        bh[ct][1] = b_hh[DD + ch];
        bh[ct][2] = b_hh[2 * DD + ch];
    }

    // Hoist all loop-invariant W_hh B-fragments into registers (384 VGPRs):
    // bf[ct][gate][kc].  gfx1250 waves address up to 1024 VGPRs.
    v16h bf[2][3][8];
#pragma unroll
    for (int ct = 0; ct < 2; ++ct) {
        const int ch = cb + ct * 16 + lh;
#pragma unroll
        for (int g = 0; g < 3; ++g) {
            const _Float16* wp = whh + (size_t)(g * DD + ch) * DD + hi * 16;
#pragma unroll
            for (int kc = 0; kc < 8; ++kc)
                bf[ct][g][kc] = *(const v16h*)(wp + kc * 32);
        }
    }

    float hbuf[4][2][8];                            // f32 recurrence state
#pragma unroll
    for (int mt = 0; mt < 4; ++mt)
#pragma unroll
        for (int ct = 0; ct < 2; ++ct)
#pragma unroll
            for (int i = 0; i < 8; ++i) hbuf[mt][ct][i] = 0.0f;

    __syncthreads();

#pragma unroll 1
    for (int j = 0; j < KWIN; ++j) {
#pragma unroll
        for (int mt = 0; mt < 4; ++mt) {
            // A-fragments for this 16-row strip (shared by both ct and 3 gates)
            v16h af[8];
#pragma unroll
            for (int kc = 0; kc < 8; ++kc) {
                const _Float16* ap = &Hh[mt * 16 + lh][kc * 32 + hi * 16];
                AFrag a;
                a.h[0] = *(const v8h*)ap;
                a.h[1] = *(const v8h*)(ap + 8);
                af[kc] = a.v;
            }
#pragma unroll
            for (int ct = 0; ct < 2; ++ct) {
                const int ch = cb + ct * 16 + lh;
                v8f accr = {}, accz = {}, accn = {};
#pragma unroll
                for (int kc = 0; kc < 8; ++kc) {
                    accr = __builtin_amdgcn_wmma_f32_16x16x32_f16(
                             false, af[kc], false, bf[ct][0][kc], (short)0, accr, false, false);
                    accz = __builtin_amdgcn_wmma_f32_16x16x32_f16(
                             false, af[kc], false, bf[ct][1][kc], (short)0, accz, false, false);
                    accn = __builtin_amdgcn_wmma_f32_16x16x32_f16(
                             false, af[kc], false, bf[ct][2][kc], (short)0, accn, false, false);
                }
                // fused gate math: all three gates of element (row, ch) in-wave
#pragma unroll
                for (int i = 0; i < 8; ++i) {
                    const int row = mt * 16 + hi * 8 + i;
                    const int p   = t0 + row + j;       // padded gi row
                    const float* gr = gib + (size_t)p * G3 + (size_t)ch * 3;
                    const float gir = gr[0];            // r/z/n contiguous (12B)
                    const float giz = gr[1];
                    const float gin = gr[2];
                    const float ghr = accr[i] + bh[ct][0];
                    const float ghz = accz[i] + bh[ct][1];
                    const float ghn = accn[i] + bh[ct][2];
                    const float r = 1.0f / (1.0f + __expf(-(gir + ghr)));
                    const float z = 1.0f / (1.0f + __expf(-(giz + ghz)));
                    const float n = tanhf(gin + r * ghn);
                    hbuf[mt][ct][i] = (1.0f - z) * n + z * hbuf[mt][ct][i];
                }
            }
        }
        __syncthreads();                    // all waves done reading Hh
#pragma unroll
        for (int mt = 0; mt < 4; ++mt)
#pragma unroll
            for (int ct = 0; ct < 2; ++ct)
#pragma unroll
                for (int i = 0; i < 8; ++i)
                    Hh[mt * 16 + hi * 8 + i][cb + ct * 16 + lh] =
                        (_Float16)hbuf[mt][ct][i];
        __syncthreads();                    // Hh updated for next step
    }

    // ---- fused residual + LayerNorm --------------------------------------
#pragma unroll
    for (int mt = 0; mt < 4; ++mt)
#pragma unroll
        for (int ct = 0; ct < 2; ++ct)
#pragma unroll
            for (int i = 0; i < 8; ++i) {
                const int row = mt * 16 + hi * 8 + i;
                const int t   = t0 + row;
                const int ch  = cb + ct * 16 + lh;
                const float y = x[((size_t)bi * LL + t) * DD + ch] + hbuf[mt][ct][i];
                hbuf[mt][ct][i] = y;        // keep y for the write phase
                float s = y, ss = y * y;
#pragma unroll
                for (int off = 1; off < 16; off <<= 1) {   // stays in 16-lane half
                    s  += __shfl_xor(s,  off, 32);
                    ss += __shfl_xor(ss, off, 32);
                }
                if (lh == 0) {                               // lanes 0 and 16
                    atomicAdd(&rsum[row],   s);              // ds_add_f32
                    atomicAdd(&rsumsq[row], ss);
                }
            }
    __syncthreads();
#pragma unroll
    for (int mt = 0; mt < 4; ++mt)
#pragma unroll
        for (int ct = 0; ct < 2; ++ct)
#pragma unroll
            for (int i = 0; i < 8; ++i) {
                const int row = mt * 16 + hi * 8 + i;
                const int t   = t0 + row;
                const int ch  = cb + ct * 16 + lh;
                const float mu  = rsum[row] * (1.0f / DD);
                const float var = rsumsq[row] * (1.0f / DD) - mu * mu;
                const float rs  = rsqrtf(var + LNEPS);
                out[((size_t)bi * LL + t) * DD + ch] =
                    (hbuf[mt][ct][i] - mu) * rs * gamma[ch] + beta[ch];
            }
}

// ---------------------------------------------------------------------------
extern "C" void kernel_launch(void* const* d_in, const int* in_sizes, int n_in,
                              void* d_out, int out_size, void* d_ws, size_t ws_size,
                              hipStream_t stream) {
    const float* x     = (const float*)d_in[0];
    const float* W_ih  = (const float*)d_in[1];
    const float* W_hh  = (const float*)d_in[2];
    const float* b_ih  = (const float*)d_in[3];
    const float* b_hh  = (const float*)d_in[4];
    const float* gamma = (const float*)d_in[5];
    const float* beta  = (const float*)d_in[6];
    float* out = (float*)d_out;

    char* ws = (char*)d_ws;
    // workspace layout (bytes):
    //   [0,       393216)  W_ih f16  (768x256)
    //   [393216,  786432)  W_hh f16  (768x256)
    //   [786432,  9240576) xpad f16  (8x2064x256, zero left-pad)
    //   [9240576, ~60MB )  gi   f32  (8x2064x768, interleaved [p][c*3+g])
    _Float16* wih_h = (_Float16*)(ws);
    _Float16* whh_h = (_Float16*)(ws + 393216);
    _Float16* xpad  = (_Float16*)(ws + 786432);
    float*    gi    = (float*)   (ws + 9240576);

    const int wn = G3 * DD;                       // 196608
    cvt_w_kernel<<<(wn + 255) / 256, 256, 0, stream>>>(W_ih, wih_h, wn);
    cvt_w_kernel<<<(wn + 255) / 256, 256, 0, stream>>>(W_hh, whh_h, wn);

    const int xn = BB * LPAD * DD;                // 4227072
    build_xpad_kernel<<<(xn + 255) / 256, 256, 0, stream>>>(x, xpad);

    gi_gemm_kernel<<<(BB * LPAD) / 64, 256, 0, stream>>>(xpad, wih_h, b_ih, gi);

    gru_ln_kernel<<<(BB * LL) / 64, 256, 0, stream>>>(whh_h, gi, x,
                                                      b_hh, gamma, beta, out);
}
